// GCN_55972013802296
// MI455X (gfx1250) — compile-verified, matrix operand order checked by reading, not run on hardware
//
#include <hip/hip_runtime.h>
#include <hip/hip_bf16.h>
#include <math.h>

#define FEATS 128
#define BN_EPS 1e-5f

typedef __attribute__((ext_vector_type(2)))  float  v2f;
typedef __attribute__((ext_vector_type(8)))  float  v8f;
typedef __attribute__((ext_vector_type(16))) __bf16 v16bf;

#if defined(__AMDGCN__) && __has_builtin(__builtin_amdgcn_wmma_f32_16x16x4_f32)
#define USE_F32_WMMA 1
#else
#define USE_F32_WMMA 0
#endif

// ---------------- degree / symmetric norm ----------------
__global__ void deg_init_kernel(float* __restrict__ deg, int n) {
    int i = blockIdx.x * blockDim.x + threadIdx.x;
    if (i < n) deg[i] = 1.0f;  // self-loop contribution
}

__global__ void deg_edges_kernel(const int* __restrict__ col, float* __restrict__ deg, int e) {
    int i = blockIdx.x * blockDim.x + threadIdx.x;
    if (i < e) atomicAdd(&deg[col[i]], 1.0f);
}

__global__ void dinv_kernel(float* __restrict__ deg, int n) {
    int i = blockIdx.x * blockDim.x + threadIdx.x;
    if (i < n) {
        float d = deg[i];
        deg[i] = (d > 0.0f) ? rsqrtf(d) : 0.0f;
    }
}

// ---------------- GEMM: H = X @ W via WMMA ----------------
// Block = 256 threads = 8 waves. Block computes rows [16*bid, 16*bid+16) x all 128 cols.
// Wave wv computes the 16x16 tile at cols [16*wv, 16*wv+16).
__global__ __launch_bounds__(256)
void gemm_wmma_kernel(const float* __restrict__ X, const float* __restrict__ W,
                      float* __restrict__ H, int n) {
    __shared__ float sX[16 * 132];  // padded stride to avoid bank conflicts

    const int tid = threadIdx.x;
    const int rowbase = blockIdx.x * 16;

    // Cooperative load of the 16x128 A-tile into LDS (8 floats / thread).
    {
        int r  = tid >> 4;          // 0..15
        int c8 = (tid & 15) * 8;    // 0..120
        int srow = rowbase + r; if (srow > n - 1) srow = n - 1;
        const float* src = X + (size_t)srow * FEATS + c8;
        float4 a0 = ((const float4*)src)[0];
        float4 a1 = ((const float4*)src)[1];
        *(float4*)&sX[r * 132 + c8]     = a0;
        *(float4*)&sX[r * 132 + c8 + 4] = a1;
    }
    __syncthreads();

    const int wv   = tid >> 5;
    const int lane = tid & 31;
    const int m    = lane & 15;
    const int half = lane >> 4;
    const int col  = wv * 16 + m;

    v8f c = {};

#if USE_F32_WMMA
    // A 16x4 fp32: VGPR0 = K(=k0)+2*half, VGPR1 = +1 ; lanes 0-15 -> M=lane, 16-31 -> M=lane-16
    #pragma unroll
    for (int k0 = 0; k0 < FEATS; k0 += 4) {
        int ka = k0 + 2 * half;
        v2f a, b;
        a.x = sX[m * 132 + ka];
        a.y = sX[m * 132 + ka + 1];
        b.x = W[(size_t)ka * FEATS + col];
        b.y = W[(size_t)(ka + 1) * FEATS + col];
        c = __builtin_amdgcn_wmma_f32_16x16x4_f32(false, a, false, b, (short)0, c,
                                                  false, false);
    }
#else
    // Fallback: bf16 K=32 WMMA (codegen-confirmed builtin), cvt on the fly.
    for (int k0 = 0; k0 < FEATS; k0 += 32) {
        v16bf a, b;
        #pragma unroll
        for (int j = 0; j < 8; ++j) {
            int kb = (j < 4) ? (k0 + 2 * j + 8 * half)
                             : (k0 + 16 + 2 * (j - 4) + 8 * half);
            a[2 * j]     = (__bf16)sX[m * 132 + kb];
            a[2 * j + 1] = (__bf16)sX[m * 132 + kb + 1];
            b[2 * j]     = (__bf16)W[(size_t)kb * FEATS + col];
            b[2 * j + 1] = (__bf16)W[(size_t)(kb + 1) * FEATS + col];
        }
        c = __builtin_amdgcn_wmma_f32_16x16x32_bf16(false, a, false, b, (short)0, c,
                                                    false, false);
    }
#endif

    // C/D layout: VGPR r -> row M=r (lanes 0-15) or M=8+r (lanes 16-31), col = lane%16.
    #pragma unroll
    for (int r = 0; r < 8; ++r) {
        int row = rowbase + r + 8 * half;
        if (row < n) H[(size_t)row * FEATS + col] = c[r];
    }
}

// ---------------- aggregation ----------------
// Init A with self-loop term + bias: A[i] = b + dinv[i]^2 * H[i]
__global__ __launch_bounds__(256)
void agg_init_kernel(const float* __restrict__ H, const float* __restrict__ dinv,
                     const float* __restrict__ bias, float* __restrict__ A, int n) {
    int gid = blockIdx.x * blockDim.x + threadIdx.x;
    if (gid >= n * 32) return;
    int node = gid >> 5, j4 = gid & 31;
    float w = dinv[node]; w *= w;
    float4 h = ((const float4*)H)[(size_t)node * 32 + j4];
    float4 b = ((const float4*)bias)[j4];
    float4 o;
    o.x = b.x + w * h.x; o.y = b.y + w * h.y;
    o.z = b.z + w * h.z; o.w = b.w + w * h.w;
    ((float4*)A)[(size_t)node * 32 + j4] = o;
}

// One wave per edge: gather H[row], scale, scatter-add into A[col]. L2-resident.
__global__ __launch_bounds__(256)
void agg_edges_kernel(const int* __restrict__ ei, const float* __restrict__ dinv,
                      const float* __restrict__ H, float* __restrict__ A, int e) {
    int gid  = blockIdx.x * blockDim.x + threadIdx.x;
    int edge = gid >> 5;
    int lane = gid & 31;
    if (edge >= e) return;
    int u = ei[edge];
    int v = ei[e + edge];
    float w = dinv[u] * dinv[v];
    float4 h = ((const float4*)H)[(size_t)u * 32 + lane];
    float* dst = A + (size_t)v * FEATS + lane * 4;
    atomicAdd(dst + 0, w * h.x);
    atomicAdd(dst + 1, w * h.y);
    atomicAdd(dst + 2, w * h.z);
    atomicAdd(dst + 3, w * h.w);
}

// ---------------- BatchNorm (training-mode batch stats) + ReLU ----------------
__global__ void zero_stats_kernel(float* __restrict__ stats) {
    stats[threadIdx.x] = 0.0f;  // launched with 256 threads (sum + sumsq)
}

__global__ __launch_bounds__(128)
void bn_stats_kernel(const float* __restrict__ A, float* __restrict__ stats,
                     int n, int rows) {
    int f  = threadIdx.x;               // feature column
    int r0 = blockIdx.x * rows;
    int r1 = r0 + rows; if (r1 > n) r1 = n;
    float s = 0.0f, q = 0.0f;
    for (int r = r0; r < r1; ++r) {
        float v = A[(size_t)r * FEATS + f];
        s += v;
        q += v * v;
    }
    atomicAdd(&stats[f], s);
    atomicAdd(&stats[FEATS + f], q);
}

__global__ __launch_bounds__(256)
void bn_apply_kernel(float* __restrict__ A, const float* __restrict__ stats,
                     const float* __restrict__ gamma, const float* __restrict__ beta,
                     int n) {
    int gid = blockIdx.x * blockDim.x + threadIdx.x;
    if (gid >= n * 32) return;
    int j4 = gid & 31;
    float inv_n = 1.0f / (float)n;
    float4 v = ((const float4*)A)[gid];
    float4 s1 = ((const float4*)stats)[j4];
    float4 s2 = ((const float4*)(stats + FEATS))[j4];
    float4 g = ((const float4*)gamma)[j4];
    float4 b = ((const float4*)beta)[j4];
    float4 o;
    {
        float mean = s1.x * inv_n, var = s2.x * inv_n - mean * mean;
        o.x = fmaxf(g.x * (v.x - mean) * rsqrtf(var + BN_EPS) + b.x, 0.0f);
    }
    {
        float mean = s1.y * inv_n, var = s2.y * inv_n - mean * mean;
        o.y = fmaxf(g.y * (v.y - mean) * rsqrtf(var + BN_EPS) + b.y, 0.0f);
    }
    {
        float mean = s1.z * inv_n, var = s2.z * inv_n - mean * mean;
        o.z = fmaxf(g.z * (v.z - mean) * rsqrtf(var + BN_EPS) + b.z, 0.0f);
    }
    {
        float mean = s1.w * inv_n, var = s2.w * inv_n - mean * mean;
        o.w = fmaxf(g.w * (v.w - mean) * rsqrtf(var + BN_EPS) + b.w, 0.0f);
    }
    ((float4*)A)[gid] = o;
}

// ---------------- edge head: sigmoid(dot(x[u]*x[v], fc_w) + fc_b) ----------------
__global__ __launch_bounds__(256)
void edge_out_kernel(const int* __restrict__ ei, const float* __restrict__ X,
                     const float* __restrict__ fw, const float* __restrict__ fb,
                     float* __restrict__ out, int e) {
    int gid  = blockIdx.x * blockDim.x + threadIdx.x;
    int edge = gid >> 5;
    int lane = gid & 31;
    if (edge >= e) return;
    int u = ei[edge];
    int v = ei[e + edge];
    float4 a = ((const float4*)X)[(size_t)u * 32 + lane];
    float4 b = ((const float4*)X)[(size_t)v * 32 + lane];
    float4 w = ((const float4*)fw)[lane];
    float p = a.x * b.x * w.x + a.y * b.y * w.y + a.z * b.z * w.z + a.w * b.w * w.w;
    #pragma unroll
    for (int off = 16; off > 0; off >>= 1)
        p += __shfl_xor(p, off, 32);
    if (lane == 0)
        out[edge] = 1.0f / (1.0f + expf(-(p + fb[0])));
}

// ---------------- host launcher ----------------
extern "C" void kernel_launch(void* const* d_in, const int* in_sizes, int n_in,
                              void* d_out, int out_size, void* d_ws, size_t ws_size,
                              hipStream_t stream) {
    const float* x      = (const float*)d_in[0];
    const int*   ei     = (const int*)d_in[1];     // [2, E] int32
    const float* conv_w = (const float*)d_in[2];   // [L,128,128]
    const float* conv_b = (const float*)d_in[3];   // [L,128]
    const float* bn_g   = (const float*)d_in[4];
    const float* bn_b   = (const float*)d_in[5];
    const float* fc_w   = (const float*)d_in[6];   // [1,128]
    const float* fc_b   = (const float*)d_in[7];   // [1]
    float* out = (float*)d_out;

    const int n = in_sizes[0] / FEATS;
    const int e = in_sizes[1] / 2;
    const int layers = in_sizes[2] / (FEATS * FEATS);

    // Workspace layout (floats): dinv[n] | A[n*128] | H[n*128] | stats[256]
    float* ws    = (float*)d_ws;
    float* dinv  = ws;
    float* A     = dinv + n;
    float* H     = A + (size_t)n * FEATS;
    float* stats = H + (size_t)n * FEATS;

    deg_init_kernel<<<(n + 255) / 256, 256, 0, stream>>>(dinv, n);
    deg_edges_kernel<<<(e + 255) / 256, 256, 0, stream>>>(ei + e, dinv, e);
    dinv_kernel<<<(n + 255) / 256, 256, 0, stream>>>(dinv, n);

    const int nv = (n * 32 + 255) / 256;   // float4-per-thread elementwise grids
    const int ew = (e + 7) / 8;            // 8 edges (waves) per 256-thread block
    const int BN_BLOCKS = 200;
    const int rows = (n + BN_BLOCKS - 1) / BN_BLOCKS;

    for (int l = 0; l < layers; ++l) {
        const float* xin = (l == 0) ? x : A;
        gemm_wmma_kernel<<<(n + 15) / 16, 256, 0, stream>>>(
            xin, conv_w + (size_t)l * FEATS * FEATS, H, n);
        agg_init_kernel<<<nv, 256, 0, stream>>>(H, dinv, conv_b + l * FEATS, A, n);
        agg_edges_kernel<<<ew, 256, 0, stream>>>(ei, dinv, H, A, e);
        zero_stats_kernel<<<1, 256, 0, stream>>>(stats);
        bn_stats_kernel<<<BN_BLOCKS, 128, 0, stream>>>(A, stats, n, rows);
        bn_apply_kernel<<<nv, 256, 0, stream>>>(A, stats, bn_g + l * FEATS,
                                                bn_b + l * FEATS, n);
    }

    edge_out_kernel<<<ew, 256, 0, stream>>>(ei, A, fc_w, fc_b, out, e);
}
